// FP4Linear_37984690766228
// MI455X (gfx1250) — matrix operand order repaired and checked by reading, not compiled, not run on hardware
//
#include <hip/hip_runtime.h>

typedef __attribute__((ext_vector_type(16))) __bf16 v16bf;
typedef __attribute__((ext_vector_type(8)))  float  v8f;
typedef int v4i __attribute__((__vector_size__(16)));

#define K_DIM 4096
#define N_DIM 4096
#define BLK_M 128
#define BLK_N 128
#define BLK_K 32
#define LDA   40     // 32 + 8 bf16 pad (16B)
#define LDB   136    // 128 + 8 bf16 pad (16B)

#if defined(__gfx1250__) &&                                         \
    __has_builtin(__builtin_amdgcn_global_load_async_to_lds_b128) && \
    __has_builtin(__builtin_amdgcn_s_wait_asynccnt)
#define USE_ASYNC 1
#else
#define USE_ASYNC 0
#endif

#if USE_ASYNC
#define GLD_ASYNC_B128(g, l)                                              \
  __builtin_amdgcn_global_load_async_to_lds_b128(                         \
      (__attribute__((address_space(1))) v4i*)(uintptr_t)(g),             \
      (__attribute__((address_space(3))) v4i*)(unsigned)(uintptr_t)(l),   \
      0, 0)
#endif

__device__ __forceinline__ unsigned f32_bf16(float f) {
  unsigned u = __float_as_uint(f);
  u = (u + 0x7FFFu + ((u >> 16) & 1u)) >> 16;
  return u & 0xFFFFu;
}

// ---------------------------------------------------------------------------
// Prepass 1: blockwise (64) absmax FP4 quant-dequant of W[o][k] (f32),
// written transposed as bf16 into Wt[k][o].
// ---------------------------------------------------------------------------
__global__ __launch_bounds__(256) void fp4_dequant_transpose(
    const float* __restrict__ W, unsigned short* __restrict__ Wt) {
  __shared__ float tile[64][68];
  __shared__ float amax[64];
  const int t  = threadIdx.x;
  const int k0 = blockIdx.x * 64;
  const int o0 = blockIdx.y * 64;

  {
    const int r = t >> 2, c = (t & 3) * 16;
    const float4* src = (const float4*)(W + (size_t)(o0 + r) * K_DIM + k0 + c);
#pragma unroll
    for (int i = 0; i < 4; ++i) {
      float4 v = src[i];
      tile[r][c + 4 * i + 0] = v.x;
      tile[r][c + 4 * i + 1] = v.y;
      tile[r][c + 4 * i + 2] = v.z;
      tile[r][c + 4 * i + 3] = v.w;
    }
  }
  __syncthreads();
  if (t < 64) {
    float m = 0.f;
#pragma unroll 8
    for (int j = 0; j < 64; ++j) m = fmaxf(m, fabsf(tile[t][j]));
    amax[t] = m;
  }
  __syncthreads();

  const float codes[16] = {0.0f,          0.0052083333f, 0.6666667f,  1.0f,
                           0.33333334f,   0.5f,          0.16666667f, 0.25f,
                           -0.0f,         -0.0052083333f,-0.6666667f, -1.0f,
                           -0.33333334f,  -0.5f,         -0.16666667f,-0.25f};
#pragma unroll
  for (int it = 0; it < 16; ++it) {
    const int idx = t + 256 * it;
    const int kk = idx >> 6, oo = idx & 63;
    const float am   = amax[oo];
    const float safe = (am == 0.f) ? 1.f : am;
    const float s    = tile[oo][kk] / safe;
    float best = 1e30f, bv = 0.f;
#pragma unroll
    for (int c = 0; c < 16; ++c) {
      const float d = fabsf(s - codes[c]);
      if (d < best) { best = d; bv = codes[c]; }
    }
    Wt[(size_t)(k0 + kk) * N_DIM + o0 + oo] =
        (unsigned short)f32_bf16(bv * am);
  }
}

// ---------------------------------------------------------------------------
// Prepass 2: x f32 -> bf16
// ---------------------------------------------------------------------------
__global__ __launch_bounds__(256) void x_to_bf16(
    const float* __restrict__ X, unsigned short* __restrict__ Xb) {
  const size_t i = ((size_t)blockIdx.x * 256 + threadIdx.x) * 8;
  const float4 a = *(const float4*)(X + i);
  const float4 b = *(const float4*)(X + i + 4);
  uint4 o;
  o.x = f32_bf16(a.x) | (f32_bf16(a.y) << 16);
  o.y = f32_bf16(a.z) | (f32_bf16(a.w) << 16);
  o.z = f32_bf16(b.x) | (f32_bf16(b.y) << 16);
  o.w = f32_bf16(b.z) | (f32_bf16(b.w) << 16);
  *(uint4*)(Xb + i) = o;
}

// ---------------------------------------------------------------------------
// bf16 WMMA GEMM: Out[m][n] = sum_k Xb[m][k] * Wt[k][n] + bias[n]
// 128x128 block tile, 8 waves (2x4), 64x32 per wave (acc = 64 VGPRs),
// K-step 32, double-buffered LDS fed by async global->LDS DMA.
// Loop is kept un-unrolled (runtime buffer pointers) so accumulators stay
// pinned and WMMA C->D coalesces in place.
// ---------------------------------------------------------------------------
__global__ __launch_bounds__(256) void fp4_gemm_bf16(
    const unsigned short* __restrict__ Xb,
    const unsigned short* __restrict__ Wt,
    const float* __restrict__ bias, float* __restrict__ Out) {
  __shared__ unsigned short sA[2 * BLK_M * LDA];  // bf16 bits, [row][k]
  __shared__ unsigned short sB[2 * BLK_K * LDB];  // bf16 bits, [k][n]

  const int t    = threadIdx.x;
  const int n0   = blockIdx.x * BLK_N;
  const int m0   = blockIdx.y * BLK_M;
  const int lane = t & 31;
  const int wave = t >> 5;
  const int wm   = wave >> 2;  // 0..1  (64 rows each)
  const int wn   = wave & 3;   // 0..3  (32 cols each)
  const int half = lane >> 4;  // 0/1
  const int l15  = lane & 15;

  // A staging: each thread moves 32B (row = t>>1, 16 bf16)
  const int aRow = t >> 1;
  const int aCol = (t & 1) * 16;
  const unsigned short* gA = Xb + (size_t)(m0 + aRow) * K_DIM + aCol;
  // B staging: each thread moves 32B (row = t>>3, 16 bf16)
  const int bRow = t >> 3;
  const int bCol = (t & 7) * 16;
  const unsigned short* gB = Wt + (size_t)bRow * N_DIM + n0 + bCol;

  v8f acc[4][2];
#pragma unroll
  for (int i = 0; i < 4; ++i)
#pragma unroll
    for (int j = 0; j < 2; ++j) {
      v8f z = {0.f, 0.f, 0.f, 0.f, 0.f, 0.f, 0.f, 0.f};
      acc[i][j] = z;
    }

  auto compute = [&](const unsigned short* sAc, const unsigned short* sBc) {
    // B frags stay live (2 x 8 VGPRs); lane&15 = N col, elements = K (table order)
    v16bf bfr[2];
#pragma unroll
    for (int ni = 0; ni < 2; ++ni) {
      const unsigned short* p = sBc + wn * 32 + ni * 16 + l15;
      union { unsigned short s[16]; v16bf v; } u;
#pragma unroll
      for (int s = 0; s < 16; ++s) {
        const int k = (s < 8) ? (half * 8 + s) : (16 + half * 8 + (s - 8));
        u.s[s] = p[k * LDB];
      }
      bfr[ni] = u.v;
    }
    // A frags transient: load -> 2 WMMAs -> dead
#pragma unroll
    for (int mi = 0; mi < 4; ++mi) {
      const unsigned short* p =
          sAc + (wm * 64 + mi * 16 + l15) * LDA + half * 8;
      union { uint4 q[2]; v16bf v; } u;
      u.q[0] = *(const uint4*)p;
      u.q[1] = *(const uint4*)(p + 16);
      const v16bf a = u.v;
      acc[mi][0] = __builtin_amdgcn_wmma_f32_16x16x32_bf16(
          false, a, false, bfr[0], (short)0, acc[mi][0], false, false);
      acc[mi][1] = __builtin_amdgcn_wmma_f32_16x16x32_bf16(
          false, a, false, bfr[1], (short)0, acc[mi][1], false, false);
    }
  };

  const int kSteps = K_DIM / BLK_K;

#if USE_ASYNC
  auto stageAsync = [&](unsigned short* laBase, unsigned short* lbBase,
                        int kt) {
    const unsigned short* ga = gA + (size_t)kt * BLK_K;
    unsigned short* la = laBase + aRow * LDA + aCol;
    GLD_ASYNC_B128(ga, la);
    GLD_ASYNC_B128(ga + 8, la + 8);
    const unsigned short* gb = gB + (size_t)kt * BLK_K * N_DIM;
    unsigned short* lb = lbBase + bRow * LDB + bCol;
    GLD_ASYNC_B128(gb, lb);
    GLD_ASYNC_B128(gb + 8, lb + 8);
  };

  stageAsync(sA, sB, 0);
  __builtin_amdgcn_s_wait_asynccnt(0);
  __syncthreads();
#pragma clang loop unroll(disable)
  for (int kt = 0; kt < kSteps; ++kt) {
    const unsigned cur = (unsigned)kt & 1u;
    unsigned short* sAc = sA + cur * (BLK_M * LDA);
    unsigned short* sBc = sB + cur * (BLK_K * LDB);
    if (kt + 1 < kSteps) {
      unsigned short* sAn = sA + (cur ^ 1u) * (BLK_M * LDA);
      unsigned short* sBn = sB + (cur ^ 1u) * (BLK_K * LDB);
      stageAsync(sAn, sBn, kt + 1);
    }
    compute(sAc, sBc);
    if (kt + 1 < kSteps) __builtin_amdgcn_s_wait_asynccnt(0);
    __syncthreads();
  }
#else
  uint4 aReg[2];
  uint4 bReg[2];
  auto loadTiles = [&](int kt) {
    const uint4* pa = (const uint4*)(gA + (size_t)kt * BLK_K);
    aReg[0] = pa[0];
    aReg[1] = pa[1];
    const uint4* pb = (const uint4*)(gB + (size_t)kt * BLK_K * N_DIM);
    bReg[0] = pb[0];
    bReg[1] = pb[1];
  };
  auto storeTiles = [&](unsigned short* laBase, unsigned short* lbBase) {
    uint4* da = (uint4*)(laBase + aRow * LDA + aCol);
    da[0] = aReg[0];
    da[1] = aReg[1];
    uint4* db = (uint4*)(lbBase + bRow * LDB + bCol);
    db[0] = bReg[0];
    db[1] = bReg[1];
  };
  loadTiles(0);
  storeTiles(sA, sB);
  __syncthreads();
#pragma clang loop unroll(disable)
  for (int kt = 0; kt < kSteps; ++kt) {
    const unsigned cur = (unsigned)kt & 1u;
    unsigned short* sAc = sA + cur * (BLK_M * LDA);
    unsigned short* sBc = sB + cur * (BLK_K * LDB);
    if (kt + 1 < kSteps) loadTiles(kt + 1);
    compute(sAc, sBc);
    if (kt + 1 < kSteps)
      storeTiles(sA + (cur ^ 1u) * (BLK_M * LDA),
                 sB + (cur ^ 1u) * (BLK_K * LDB));
    __syncthreads();
  }
#endif

  // Epilogue: C/D layout -> VGPR r = M row (lanes>=16: +8), lane&15 = N col
#pragma unroll
  for (int mi = 0; mi < 4; ++mi) {
#pragma unroll
    for (int ni = 0; ni < 2; ++ni) {
      const int col = n0 + wn * 32 + ni * 16 + l15;
      const float bv = bias[col];
      const int rowBase = m0 + wm * 64 + mi * 16 + half * 8;
      float* po = Out + (size_t)rowBase * N_DIM + col;
#pragma unroll
      for (int r = 0; r < 8; ++r)
        po[(size_t)r * N_DIM] = acc[mi][ni][r] + bv;
    }
  }
}

extern "C" void kernel_launch(void* const* d_in, const int* in_sizes, int n_in,
                              void* d_out, int out_size, void* d_ws,
                              size_t ws_size, hipStream_t stream) {
  const float* x    = (const float*)d_in[0];
  const float* w    = (const float*)d_in[1];
  const float* bias = (const float*)d_in[2];
  float* out        = (float*)d_out;

  unsigned short* wt = (unsigned short*)d_ws;                    // 32 MB
  unsigned short* xb = wt + (size_t)K_DIM * N_DIM;               // 64 MB

  const int M = in_sizes[0] / K_DIM;  // 8192

  fp4_dequant_transpose<<<dim3(K_DIM / 64, N_DIM / 64), 256, 0, stream>>>(w, wt);
  x_to_bf16<<<dim3((unsigned)((size_t)M * K_DIM / 2048)), 256, 0, stream>>>(x, xb);
  fp4_gemm_bf16<<<dim3(N_DIM / BLK_N, M / BLK_M), 256, 0, stream>>>(xb, wt, bias,
                                                                    out);
}
